// SchNetLikeModel_23914377904249
// MI455X (gfx1250) — matrix operand order repaired and checked by reading, not compiled, not run on hardware
//
#include <hip/hip_runtime.h>

// ---------------------------------------------------------------------------
// SchNet-like GNN for MI455X (gfx1250).
//  - per-edge MLP hoisted to per-node MLP (identical math)
//  - GEMMs via v_wmma_f32_16x16x32_f16 with pre-packed f16 weight fragments
//  - activations kept in f32 (residual/BN) + f16 shadow (WMMA A operand)
//  - scatter/pool via vectorized loads + f32 atomics (all L2-resident)
// ---------------------------------------------------------------------------

typedef _Float16 v16h __attribute__((ext_vector_type(16)));
typedef _Float16 v8h  __attribute__((ext_vector_type(8)));
typedef float    v8f  __attribute__((ext_vector_type(8)));

#define N_NODES   50000
#define N_FEAT    128
#define HID       64
#define N_EDGES   600000
#define N_GRAPHS  256
#define BN_EPS    1e-5f
#define WPB       5          // waves per block; 3125 strips = 625 blocks * 5 waves

__device__ __forceinline__ v8f wmma_f16(v16h a, v16h b, v8f c) {
  return __builtin_amdgcn_wmma_f32_16x16x32_f16(false, a, false, b, (short)0, c,
                                                false, false);
}

__device__ __forceinline__ v16h cat8(v8h lo, v8h hi) {
  return __builtin_shufflevector(lo, hi, 0, 1, 2, 3, 4, 5, 6, 7,
                                 8, 9, 10, 11, 12, 13, 14, 15);
}

// Pack W (5 layers of KxN f32, row-major) into per-lane WMMA B fragments (f16).
// frag id within layer = nt*(K/32) + ks ; each lane stores 16 contiguous halfs:
// element e -> W[ks*32 + (lane>=16)*8 + e + (e>=8)*8][nt*16 + (lane&15)]
__global__ void pack_weights_kernel(const float* __restrict__ W,
                                    _Float16* __restrict__ P, int K, int N) {
  const int lane = threadIdx.x;            // 32 threads per block
  const int ksteps = K / 32;
  const int fragsPerLayer = (N / 16) * ksteps;
  const int fid = blockIdx.x;              // over layers*frags
  const int l = fid / fragsPerLayer;
  const int f = fid % fragsPerLayer;
  const int nt = f / ksteps;
  const int ks = f % ksteps;
  const int col = nt * 16 + (lane & 15);
  const int kb  = ks * 32 + (lane >> 4) * 8;
  const float* Wl = W + (size_t)l * K * N;
  _Float16* out = P + ((size_t)fid * 32 + lane) * 16;
#pragma unroll
  for (int e = 0; e < 16; ++e) out[e] = (_Float16)Wl[(size_t)(kb + e + (e >= 8 ? 8 : 0)) * N + col];
}

// f32 -> f16 row-major copy (for layer-0 activations), 4 elems/thread
__global__ void cvt_f16_kernel(const float* __restrict__ in,
                               _Float16* __restrict__ out) {
  const size_t i = ((size_t)blockIdx.x * blockDim.x + threadIdx.x) * 4;
#pragma unroll
  for (int j = 0; j < 4; ++j) out[i + j] = (_Float16)in[i + j];
}

// One wave: T[strip] = relu(H[strip]@W1+b1)@W2+b2 for a 16-node strip.
// Also initializes the scatter accumulator with T (self-loop term).
__global__ __launch_bounds__(WPB * 32)
void mlp_kernel(const _Float16* __restrict__ H16,
                const v16h* __restrict__ W1p, const float* __restrict__ b1,
                const v16h* __restrict__ W2p, const float* __restrict__ b2,
                float* __restrict__ T, float* __restrict__ outacc) {
  __shared__ _Float16 zbuf[WPB][16][HID];   // per-wave 16x64 hidden tile (f16)

  const int wid   = threadIdx.x >> 5;
  const int lane  = threadIdx.x & 31;
  const int m     = lane & 15;
  const int hi    = lane >> 4;
  const int node0 = (blockIdx.x * WPB + wid) * 16;

  // ---- Stage 1: Z = relu(H[16x128] @ W1[128x64] + b1) -> LDS (f16) ----
  v16h afrag[4];
#pragma unroll
  for (int ks = 0; ks < 4; ++ks) {
    const _Float16* hrow = H16 + (size_t)(node0 + m) * N_FEAT + ks * 32 + hi * 8;
    afrag[ks] = cat8(*(const v8h*)hrow, *(const v8h*)(hrow + 16));
  }

#pragma unroll
  for (int nt = 0; nt < 4; ++nt) {
    const int col = nt * 16 + m;
    v8f acc = {};
#pragma unroll
    for (int ks = 0; ks < 4; ++ks) {
      acc = wmma_f16(afrag[ks], W1p[(nt * 4 + ks) * 32 + lane], acc);
    }
    const float bias = b1[col];
#pragma unroll
    for (int v = 0; v < 8; ++v) {
      float z = acc[v] + bias;
      z = z > 0.f ? z : 0.f;
      zbuf[wid][v + 8 * hi][col] = (_Float16)z;
    }
  }

  __syncthreads();   // cross-lane LDS handoff (all waves reach this)

  // ---- Stage 2: T = Z[16x64] @ W2[64x128] + b2 ----
  v16h a2[2];
#pragma unroll
  for (int ks = 0; ks < 2; ++ks) {
    const _Float16* zr = &zbuf[wid][m][ks * 32 + hi * 8];
    a2[ks] = cat8(*(const v8h*)zr, *(const v8h*)(zr + 16));
  }

#pragma unroll
  for (int nt = 0; nt < 8; ++nt) {
    const int col = nt * 16 + m;
    v8f acc = {};
#pragma unroll
    for (int ks = 0; ks < 2; ++ks) {
      acc = wmma_f16(a2[ks], W2p[(nt * 2 + ks) * 32 + lane], acc);
    }
    const float bias = b2[col];
#pragma unroll
    for (int v = 0; v < 8; ++v) {
      const size_t idx = (size_t)(node0 + v + 8 * hi) * N_FEAT + col;
      const float val  = acc[v] + bias;
      T[idx]      = val;
      outacc[idx] = val;                    // self-loop: out[v] starts at T[v]
    }
  }
}

// out[dst] += T[src]; thread = (edge, 4 features)
__global__ void scatter_kernel(const float* __restrict__ T,
                               float* __restrict__ outacc,
                               const int* __restrict__ src,
                               const int* __restrict__ dst) {
  const long long t = (long long)blockIdx.x * blockDim.x + threadIdx.x;
  const int e = (int)(t >> 5);
  const int f = (int)(t & 31) * 4;
  if (e < N_EDGES) {
    const int s = src[e];
    const int d = dst[e];
    const float4 v = *(const float4*)&T[(size_t)s * N_FEAT + f];
    float* o = &outacc[(size_t)d * N_FEAT + f];
    atomicAdd(o + 0, v.x);
    atomicAdd(o + 1, v.y);
    atomicAdd(o + 2, v.z);
    atomicAdd(o + 3, v.w);
  }
}

// h_out = relu(BN(acc)) (+ h_in residual); also emits f16 shadow for next WMMA.
// In-place safe (pure elementwise).
__global__ void bn_relu_res_kernel(const float* __restrict__ acc,
                                   const float* __restrict__ hin,
                                   float* __restrict__ hout,
                                   _Float16* __restrict__ hout16,
                                   const float* __restrict__ gam,
                                   const float* __restrict__ bet,
                                   const float* __restrict__ mu,
                                   const float* __restrict__ var,
                                   int addres) {
  const size_t i = (size_t)blockIdx.x * blockDim.x + threadIdx.x;
  const int f = (int)(i & 127);
  float y = (acc[i] - mu[f]) * rsqrtf(var[f] + BN_EPS) * gam[f] + bet[f];
  y = fmaxf(y, 0.f);
  if (addres) y += hin[i];
  hout[i]   = y;
  hout16[i] = (_Float16)y;
}

__global__ void zero_kernel(float* __restrict__ p, int n) {
  const int i = blockIdx.x * blockDim.x + threadIdx.x;
  if (i < n) p[i] = 0.f;
}

// thread = (node, 4 features)
__global__ void pool_kernel(const float* __restrict__ h,
                            const int* __restrict__ batch,
                            float* __restrict__ sums,
                            float* __restrict__ counts) {
  const size_t t = (size_t)blockIdx.x * blockDim.x + threadIdx.x;
  const int v = (int)(t >> 5);
  const int f = (int)(t & 31) * 4;
  const int g = batch[v];
  const float4 x = *(const float4*)&h[(size_t)v * N_FEAT + f];
  float* o = &sums[(size_t)g * N_FEAT + f];
  atomicAdd(o + 0, x.x);
  atomicAdd(o + 1, x.y);
  atomicAdd(o + 2, x.z);
  atomicAdd(o + 3, x.w);
  if (f == 0) atomicAdd(&counts[g], 1.f);
}

__global__ void final_kernel(const float* __restrict__ sums,
                             const float* __restrict__ counts,
                             const float* __restrict__ lw,
                             const float* __restrict__ lb,
                             float* __restrict__ out) {
  const int g = threadIdx.x;
  float acc = 0.f;
#pragma unroll 4
  for (int f = 0; f < N_FEAT; ++f) acc += sums[(size_t)g * N_FEAT + f] * lw[f];
  float c = counts[g];
  c = c > 1.f ? c : 1.f;
  out[g] = acc / c + lb[0];
}

extern "C" void kernel_launch(void* const* d_in, const int* in_sizes, int n_in,
                              void* d_out, int out_size, void* d_ws, size_t ws_size,
                              hipStream_t stream) {
  const float* x     = (const float*)d_in[0];
  const int*   ei    = (const int*)d_in[1];     // (2, E)
  const int*   batch = (const int*)d_in[2];
  const float* W1s   = (const float*)d_in[3];   // (5,128,64)
  const float* b1s   = (const float*)d_in[4];   // (5,64)
  const float* W2s   = (const float*)d_in[5];   // (5,64,128)
  const float* b2s   = (const float*)d_in[6];   // (5,128)
  const float* gam   = (const float*)d_in[7];
  const float* bet   = (const float*)d_in[8];
  const float* mu    = (const float*)d_in[9];
  const float* var   = (const float*)d_in[10];
  const float* lw    = (const float*)d_in[11];  // (128,1)
  const float* lb    = (const float*)d_in[12];  // (1,)
  float* out = (float*)d_out;

  char* ws = (char*)d_ws;
  const size_t NF   = (size_t)N_NODES * N_FEAT;
  const size_t NFB  = NF * sizeof(float);                  // 25.6 MB
  const size_t NFB2 = NF * sizeof(_Float16);               // 12.8 MB
  float*    T      = (float*)(ws);
  float*    accb   = (float*)(ws + NFB);
  float*    hbuf   = (float*)(ws + 2 * NFB);
  _Float16* h16    = (_Float16*)(ws + 3 * NFB);
  _Float16* w1pack = (_Float16*)(ws + 3 * NFB + NFB2);               // 80 KB
  _Float16* w2pack = (_Float16*)(ws + 3 * NFB + NFB2 + 5 * 16 * 32 * 16 * 2);
  float*    sums   = (float*)(ws + 3 * NFB + NFB2 + 2 * 5 * 16 * 32 * 16 * 2);
  float*    counts = sums + (size_t)N_GRAPHS * N_FEAT;

  const int* src = ei;
  const int* dst = ei + N_EDGES;

  const int mlp_blocks     = N_NODES / (16 * WPB);               // 625
  const int scatter_blocks = (N_EDGES * 32) / 256;               // 75000
  const int elem_blocks    = (int)(NF / 256);                    // 25000
  const int pool_blocks    = (N_NODES * 32) / 256;               // 6250
  const int cvt_blocks     = (int)(NF / (256 * 4));              // 6250

  // One-time per launch: pack weights to f16 fragments, shadow x in f16.
  pack_weights_kernel<<<5 * 16, 32, 0, stream>>>(W1s, w1pack, N_FEAT, HID);
  pack_weights_kernel<<<5 * 16, 32, 0, stream>>>(W2s, w2pack, HID, N_FEAT);
  cvt_f16_kernel<<<cvt_blocks, 256, 0, stream>>>(x, h16);

  const float* hin = x;
  for (int l = 0; l < 5; ++l) {
    mlp_kernel<<<mlp_blocks, WPB * 32, 0, stream>>>(
        h16,
        (const v16h*)(w1pack + (size_t)l * 16 * 32 * 16), b1s + (size_t)l * HID,
        (const v16h*)(w2pack + (size_t)l * 16 * 32 * 16), b2s + (size_t)l * N_FEAT,
        T, accb);
    scatter_kernel<<<scatter_blocks, 256, 0, stream>>>(T, accb, src, dst);
    bn_relu_res_kernel<<<elem_blocks, 256, 0, stream>>>(
        accb, hin, hbuf, h16,
        gam + (size_t)l * N_FEAT, bet + (size_t)l * N_FEAT,
        mu + (size_t)l * N_FEAT, var + (size_t)l * N_FEAT,
        (l > 0) ? 1 : 0);
    hin = hbuf;
  }

  const int nz = N_GRAPHS * N_FEAT + N_GRAPHS;
  zero_kernel<<<(nz + 255) / 256, 256, 0, stream>>>(sums, nz);
  pool_kernel<<<pool_blocks, 256, 0, stream>>>(hbuf, batch, sums, counts);
  final_kernel<<<1, N_GRAPHS, 0, stream>>>(sums, counts, lw, lb, out);
}
